// GnBlock_46231027974467
// MI455X (gfx1250) — matrix operand order repaired and checked by reading, not compiled, not run on hardware
//
#include <hip/hip_runtime.h>

#define N_NODES 100000
#define N_EDGES 600000
#define H       128
#define HALF    64
#define KE      384   // 3*H  (edge MLP fan-in)
#define KN      192   // H + H/2 (node MLP fan-in)
#define LN_EPS  1e-5f

typedef __attribute__((ext_vector_type(16))) __bf16 v16bf;
typedef __attribute__((ext_vector_type(8)))  __bf16 v8bf;
typedef __attribute__((ext_vector_type(8)))  float  v8f;

union FragA { v16bf v; v8bf h[2]; };

__device__ __forceinline__ unsigned short f2bf(float f) {
  union { float f; unsigned int u; } c; c.f = f;
  unsigned int u = c.u;
  if ((u & 0x7F800000u) != 0x7F800000u)        // finite: round-to-nearest-even
    u += 0x7FFFu + ((u >> 16) & 1u);
  return (unsigned short)(u >> 16);
}

// ---------------------------------------------------------------------------
// Wave-level GEMM strip: 16 rows x 128 cols, A (bf16) in LDS, B pre-swizzled
// fragments in global.  acc[nt] is the 16x16 fp32 C/D tile for n-tile nt.
//
// A-fragment layout (ISA 16-bit A 16x32): lane<16 row=lane, K = k0..k0+7 and
// k0+16..k0+23 with k0 = 32*kt; lane>=16 same rows, +8 K offsets.
// B-fragment layout: lane<16 col=lane, K=0..7,16..23; lane>=16 col=lane-16, K=8..15,24..31
// (pre-packed by kconv so each (kt,nt) frag is 32 lanes x 16 bf16 contiguous).
// ---------------------------------------------------------------------------
__device__ __forceinline__ void gemm_strip(const unsigned short* A, int astride, int ktc,
                                           const unsigned short* __restrict__ wf,
                                           int lane, v8f* acc) {
  const int rl = lane & 15, hi = lane >> 4;
  const unsigned short* arow = A + rl * astride;
  for (int kt = 0; kt < ktc; ++kt) {
    FragA a;
    const int k0 = kt * 32 + hi * 8;
    a.h[0] = *(const v8bf*)(arow + k0);
    a.h[1] = *(const v8bf*)(arow + k0 + 16);
    const unsigned short* wk = wf + (((kt * 8) * 32 + lane) << 4);
#pragma unroll
    for (int nt = 0; nt < 8; ++nt) {
      v16bf b = *(const v16bf*)(wk + (nt << 9));   // nt * 32 lanes * 16 elems
      acc[nt] = __builtin_amdgcn_wmma_f32_16x16x32_bf16(
          false, a.v, false, b, (short)0, acc[nt], false, false);
    }
  }
}

// bias + SiLU, store bf16 into per-wave LDS hidden buffer (row-major).
// D layout: VGPR i, lane<16 -> row i, col=lane ; lane>=16 -> row 8+i, col=lane-16.
__device__ __forceinline__ void silu_store(const v8f* acc, const float* __restrict__ bias,
                                           unsigned short* hb, int hstride, int lane) {
  const int rl = lane & 15, hi = lane >> 4;
#pragma unroll
  for (int nt = 0; nt < 8; ++nt) {
    const int col = nt * 16 + rl;
    const float bv = bias[col];
#pragma unroll
    for (int i = 0; i < 8; ++i) {
      float v = acc[nt][i] + bv;
      v = v * (1.0f / (1.0f + __expf(-v)));
      hb[(hi * 8 + i) * hstride + col] = f2bf(v);
    }
  }
}

// bias add + LayerNorm(g,beta) in place.  Row (hi*8+i) lives in the 16 lanes of
// half 'hi'; xor masks 1..8 reduce within that half.
__device__ __forceinline__ void bias_ln_inplace(v8f* acc, const float* __restrict__ bias,
                                                const float* __restrict__ g,
                                                const float* __restrict__ beta, int lane) {
  const int rl = lane & 15;
  float s[8], q[8];
#pragma unroll
  for (int i = 0; i < 8; ++i) { s[i] = 0.f; q[i] = 0.f; }
#pragma unroll
  for (int nt = 0; nt < 8; ++nt) {
    const float bv = bias[nt * 16 + rl];
#pragma unroll
    for (int i = 0; i < 8; ++i) {
      float v = acc[nt][i] + bv;
      acc[nt][i] = v;
      s[i] += v; q[i] += v * v;
    }
  }
#pragma unroll
  for (int i = 0; i < 8; ++i) {
#pragma unroll
    for (int m = 1; m <= 8; m <<= 1) {
      s[i] += __shfl_xor(s[i], m, 32);
      q[i] += __shfl_xor(q[i], m, 32);
    }
    const float mean = s[i] * (1.0f / 128.0f);
    const float var  = q[i] * (1.0f / 128.0f) - mean * mean;
    s[i] = mean;
    q[i] = rsqrtf(var + LN_EPS);
  }
#pragma unroll
  for (int nt = 0; nt < 8; ++nt) {
    const int col = nt * 16 + rl;
    const float gv = g[col], bb = beta[col];
#pragma unroll
    for (int i = 0; i < 8; ++i)
      acc[nt][i] = (acc[nt][i] - s[i]) * q[i] * gv + bb;
  }
}

// ---------------------------------------------------------------------------
// Weight fp32 -> bf16 B-fragment pre-swizzle.  dst index i decodes to
// (kt, nt, lane, e); source element W[k][n] with k = 32*kt + 8*(lane>>4) + (e&7) + 16*(e>>3),
// n = 16*nt + (lane&15).
// ---------------------------------------------------------------------------
__global__ void kconv(const float* __restrict__ W, unsigned short* __restrict__ dst, int total) {
  const int i = blockIdx.x * 256 + threadIdx.x;
  if (i >= total) return;
  const int e = i & 15, lane = (i >> 4) & 31, nt = (i >> 9) & 7, kt = i >> 12;
  const int k = kt * 32 + (lane >> 4) * 8 + (e & 7) + ((e >> 3) << 4);
  const int n = nt * 16 + (lane & 15);
  dst[i] = f2bf(W[k * H + n]);
}

__global__ void kzero(float* __restrict__ p, int n) {
  const int i = blockIdx.x * 256 + threadIdx.x;
  if (i < n) p[i] = 0.f;
}

// ---------------------------------------------------------------------------
// Edge block: 64 edges / WG, 128 threads (4 waves, 16 rows each).
// ---------------------------------------------------------------------------
__global__ __launch_bounds__(128)
void edge_kernel(const float* __restrict__ x, const float* __restrict__ ea,
                 const int* __restrict__ ei,
                 const unsigned short* __restrict__ w1, const unsigned short* __restrict__ w2,
                 const unsigned short* __restrict__ w3,
                 const float* __restrict__ b1, const float* __restrict__ b2,
                 const float* __restrict__ b3,
                 const float* __restrict__ g, const float* __restrict__ beta,
                 float* __restrict__ agg, float* __restrict__ out_e) {
  __shared__ unsigned short Ain[64 * 392];          // 64 rows x 384 (+8 pad) bf16
  __shared__ unsigned short Hb[4 * 2 * 16 * 136];   // per-wave double hidden buf
  const int tid = threadIdx.x;
  const int lane = tid & 31, wave = tid >> 5;
  const int tile = blockIdx.x * 64;

  // stage concat(edge_attr, x[senders], x[receivers]) -> bf16 LDS
  for (int it = tid; it < 64 * 96; it += 128) {
    const int r = it / 96, c4 = it - r * 96;
    const int ge = tile + r;
    float4 v = make_float4(0.f, 0.f, 0.f, 0.f);
    if (ge < N_EDGES) {
      if (c4 < 32)      v = ((const float4*)(ea + (size_t)ge * H))[c4];
      else if (c4 < 64) v = ((const float4*)(x + (size_t)ei[ge] * H))[c4 - 32];
      else              v = ((const float4*)(x + (size_t)ei[N_EDGES + ge] * H))[c4 - 64];
    }
    unsigned short* p = &Ain[r * 392 + c4 * 4];
    p[0] = f2bf(v.x); p[1] = f2bf(v.y); p[2] = f2bf(v.z); p[3] = f2bf(v.w);
  }
  __syncthreads();

  unsigned short* h0 = &Hb[wave * (2 * 16 * 136)];
  unsigned short* h1 = h0 + 16 * 136;
  const v8f vzero = {0.f, 0.f, 0.f, 0.f, 0.f, 0.f, 0.f, 0.f};
  v8f acc[8];

#pragma unroll
  for (int nt = 0; nt < 8; ++nt) acc[nt] = vzero;
  gemm_strip(&Ain[(wave * 16) * 392], 392, 12, w1, lane, acc);
  silu_store(acc, b1, h0, 136, lane);
  __syncthreads();

#pragma unroll
  for (int nt = 0; nt < 8; ++nt) acc[nt] = vzero;
  gemm_strip(h0, 136, 4, w2, lane, acc);
  silu_store(acc, b2, h1, 136, lane);
  __syncthreads();

#pragma unroll
  for (int nt = 0; nt < 8; ++nt) acc[nt] = vzero;
  gemm_strip(h1, 136, 4, w3, lane, acc);

  bias_ln_inplace(acc, b3, g, beta, lane);   // acc now = new_e (pre-residual)

  const int rl = lane & 15, hi = lane >> 4;
#pragma unroll
  for (int i = 0; i < 8; ++i) {
    const int ge = tile + wave * 16 + hi * 8 + i;
    if (ge < N_EDGES) {
      const int snd = ei[ge], rcv = ei[N_EDGES + ge];
#pragma unroll
      for (int nt = 0; nt < 8; ++nt) {
        const int col = nt * 16 + rl;
        const float y = acc[nt][i];
        out_e[(size_t)ge * H + col] = y + ea[(size_t)ge * H + col];
        if (nt < 4) atomicAdd(&agg[(size_t)rcv * HALF + col], y);
        else        atomicAdd(&agg[(size_t)snd * HALF + (col - HALF)], y);
      }
    }
  }
}

// ---------------------------------------------------------------------------
// Node block: 64 nodes / WG, 128 threads.
// ---------------------------------------------------------------------------
__global__ __launch_bounds__(128)
void node_kernel(const float* __restrict__ x, const float* __restrict__ agg,
                 const unsigned short* __restrict__ w1, const unsigned short* __restrict__ w2,
                 const unsigned short* __restrict__ w3,
                 const float* __restrict__ b1, const float* __restrict__ b2,
                 const float* __restrict__ b3,
                 const float* __restrict__ g, const float* __restrict__ beta,
                 float* __restrict__ out_x) {
  __shared__ unsigned short Ain[64 * 200];          // 64 rows x 192 (+8 pad) bf16
  __shared__ unsigned short Hb[4 * 2 * 16 * 136];
  const int tid = threadIdx.x;
  const int lane = tid & 31, wave = tid >> 5;
  const int tile = blockIdx.x * 64;

  // stage concat(x, agg) -> bf16 LDS
  for (int it = tid; it < 64 * 48; it += 128) {
    const int r = it / 48, c4 = it - r * 48;
    const int nd = tile + r;
    float4 v = make_float4(0.f, 0.f, 0.f, 0.f);
    if (nd < N_NODES) {
      if (c4 < 32) v = ((const float4*)(x + (size_t)nd * H))[c4];
      else         v = ((const float4*)(agg + (size_t)nd * HALF))[c4 - 32];
    }
    unsigned short* p = &Ain[r * 200 + c4 * 4];
    p[0] = f2bf(v.x); p[1] = f2bf(v.y); p[2] = f2bf(v.z); p[3] = f2bf(v.w);
  }
  __syncthreads();

  unsigned short* h0 = &Hb[wave * (2 * 16 * 136)];
  unsigned short* h1 = h0 + 16 * 136;
  const v8f vzero = {0.f, 0.f, 0.f, 0.f, 0.f, 0.f, 0.f, 0.f};
  v8f acc[8];

#pragma unroll
  for (int nt = 0; nt < 8; ++nt) acc[nt] = vzero;
  gemm_strip(&Ain[(wave * 16) * 200], 200, 6, w1, lane, acc);
  silu_store(acc, b1, h0, 136, lane);
  __syncthreads();

#pragma unroll
  for (int nt = 0; nt < 8; ++nt) acc[nt] = vzero;
  gemm_strip(h0, 136, 4, w2, lane, acc);
  silu_store(acc, b2, h1, 136, lane);
  __syncthreads();

#pragma unroll
  for (int nt = 0; nt < 8; ++nt) acc[nt] = vzero;
  gemm_strip(h1, 136, 4, w3, lane, acc);

  bias_ln_inplace(acc, b3, g, beta, lane);   // acc = new_x (pre-residual)

  const int rl = lane & 15, hi = lane >> 4;
#pragma unroll
  for (int i = 0; i < 8; ++i) {
    const int nd = tile + wave * 16 + hi * 8 + i;
    if (nd < N_NODES) {
#pragma unroll
      for (int nt = 0; nt < 8; ++nt) {
        const int col = nt * 16 + rl;
        out_x[(size_t)nd * H + col] = acc[nt][i] + x[(size_t)nd * H + col];
      }
    }
  }
}

// ---------------------------------------------------------------------------
extern "C" void kernel_launch(void* const* d_in, const int* in_sizes, int n_in,
                              void* d_out, int out_size, void* d_ws, size_t ws_size,
                              hipStream_t stream) {
  (void)in_sizes; (void)n_in; (void)out_size; (void)ws_size;
  const float* x   = (const float*)d_in[0];
  const float* ea  = (const float*)d_in[1];
  const int*   ei  = (const int*)  d_in[2];
  const float* eW1 = (const float*)d_in[3];
  const float* eb1 = (const float*)d_in[4];
  const float* eW2 = (const float*)d_in[5];
  const float* eb2 = (const float*)d_in[6];
  const float* eW3 = (const float*)d_in[7];
  const float* eb3 = (const float*)d_in[8];
  const float* e_g = (const float*)d_in[9];
  const float* e_b = (const float*)d_in[10];
  const float* nW1 = (const float*)d_in[11];
  const float* nb1 = (const float*)d_in[12];
  const float* nW2 = (const float*)d_in[13];
  const float* nb2 = (const float*)d_in[14];
  const float* nW3 = (const float*)d_in[15];
  const float* nb3 = (const float*)d_in[16];
  const float* n_g = (const float*)d_in[17];
  const float* n_b = (const float*)d_in[18];

  char* ws = (char*)d_ws;
  float* agg = (float*)ws;
  size_t off = (size_t)N_NODES * HALF * sizeof(float);      // 25.6 MB, 256B aligned
  unsigned short* eW1f = (unsigned short*)(ws + off); off += (size_t)KE * H * 2;
  unsigned short* eW2f = (unsigned short*)(ws + off); off += (size_t)H  * H * 2;
  unsigned short* eW3f = (unsigned short*)(ws + off); off += (size_t)H  * H * 2;
  unsigned short* nW1f = (unsigned short*)(ws + off); off += (size_t)KN * H * 2;
  unsigned short* nW2f = (unsigned short*)(ws + off); off += (size_t)H  * H * 2;
  unsigned short* nW3f = (unsigned short*)(ws + off); off += (size_t)H  * H * 2;

  float* out_x = (float*)d_out;
  float* out_e = out_x + (size_t)N_NODES * H;

  kconv<<<(KE * H) / 256, 256, 0, stream>>>(eW1, eW1f, KE * H);
  kconv<<<(H  * H) / 256, 256, 0, stream>>>(eW2, eW2f, H * H);
  kconv<<<(H  * H) / 256, 256, 0, stream>>>(eW3, eW3f, H * H);
  kconv<<<(KN * H) / 256, 256, 0, stream>>>(nW1, nW1f, KN * H);
  kconv<<<(H  * H) / 256, 256, 0, stream>>>(nW2, nW2f, H * H);
  kconv<<<(H  * H) / 256, 256, 0, stream>>>(nW3, nW3f, H * H);

  kzero<<<(N_NODES * HALF + 255) / 256, 256, 0, stream>>>(agg, N_NODES * HALF);

  edge_kernel<<<(N_EDGES + 63) / 64, 128, 0, stream>>>(
      x, ea, ei, eW1f, eW2f, eW3f, eb1, eb2, eb3, e_g, e_b, agg, out_e);

  node_kernel<<<(N_NODES + 63) / 64, 128, 0, stream>>>(
      x, agg, nW1f, nW2f, nW3f, nb1, nb2, nb3, n_g, n_b, out_x);
}
